// Attention_1047972020931
// MI455X (gfx1250) — matrix operand order repaired and checked by reading, not compiled, not run on hardware
//
#include <hip/hip_runtime.h>
#include <hip/hip_bf16.h>

typedef __attribute__((ext_vector_type(2)))  float  v2f;
typedef __attribute__((ext_vector_type(8)))  float  v8f;
typedef __attribute__((ext_vector_type(4)))  int    v4i;
typedef __attribute__((ext_vector_type(16))) __bf16 v16bf;

typedef v4i __attribute__((address_space(1)))* gv4i_p;  // global int4*
typedef v4i __attribute__((address_space(3)))* lv4i_p;  // LDS int4*

#define B_   2
#define S_   2048
#define NX_  1024
#define NH_  16
#define HD_  64
#define K3   (3*NX_)          // 3072
#define MROWS (B_*S_)         // 4096
#define PRES ((size_t)B_*NH_*S_*HD_)   // 4,194,304
#define OUT_ELEMS ((size_t)MROWS*NX_)  // 4,194,304
#define SCALE_LORA 8.0f

#if __has_builtin(__builtin_amdgcn_global_load_async_to_lds_b128)
#define HAVE_ASYNC_LDS 1
#endif

#if defined(HAVE_ASYNC_LDS)
#if __has_builtin(__builtin_amdgcn_s_wait_asynccnt)
#define WAIT_ASYNC(n) __builtin_amdgcn_s_wait_asynccnt(n)
#else
#define WAIT_ASYNC(n) asm volatile("s_wait_asynccnt " #n ::: "memory")
#endif
#endif

// ---------------------------------------------------------------------------
// Kernel A: fold rank-4 LoRA into the QKV weight matrix.
// W_eff[i][j] = W_attn[i][j] (+ SCALING * sum_r A[r][i]*B[j][r] on q/v cols)
// ---------------------------------------------------------------------------
__global__ __launch_bounds__(256)
void build_weff(const float* __restrict__ Wa, const float* __restrict__ lA,
                const float* __restrict__ lB, float* __restrict__ We) {
  int idx = blockIdx.x * 256 + threadIdx.x;
  if (idx >= NX_ * K3) return;
  int i = idx / K3;
  int j = idx - i * K3;
  float w = Wa[idx];
  if (j < NX_) {                       // q columns: lora_A rows 0..3, lora_B[:NX]
    float acc = 0.f;
#pragma unroll
    for (int r = 0; r < 4; ++r) acc += lA[r * NX_ + i] * lB[j * 4 + r];
    w += SCALE_LORA * acc;
  } else if (j >= 2 * NX_) {           // v columns: lora_A rows 4..7, lora_B[NX:]
    int jj = j - 2 * NX_;
    float acc = 0.f;
#pragma unroll
    for (int r = 0; r < 4; ++r) acc += lA[(4 + r) * NX_ + i] * lB[(NX_ + jj) * 4 + r];
    w += SCALE_LORA * acc;
  }
  We[idx] = w;
}

// ---------------------------------------------------------------------------
// Kernel B: QKV GEMM, fp32 WMMA (16x16x4), wave tile = 32 rows x 64 cols
// (two A fragments share four B fragments). Epilogue scatters q -> q_ws
// (b,h,s,d) and k/v -> the present region of d_out.
// ---------------------------------------------------------------------------
__global__ __launch_bounds__(256)
void qkv_gemm(const float* __restrict__ X, const float* __restrict__ We,
              const float* __restrict__ bias, float* __restrict__ qws,
              float* __restrict__ kout, float* __restrict__ vout) {
  const int lane = threadIdx.x & 31;
  const int warp = threadIdx.x >> 5;
  const int half = lane >> 4, lr = lane & 15;
  const int m0 = blockIdx.y * 64 + (warp >> 2) * 32;
  const int n0 = blockIdx.x * 256 + (warp & 3) * 64;

  v8f acc[2][4] = {};
  const float* arow0 = X + (size_t)(m0 + lr) * NX_ + 2 * half;
  const float* arow1 = arow0 + (size_t)16 * NX_;
  const float* bcol  = We + n0 + lr;

  for (int k = 0; k < NX_; k += 4) {
    v2f a0, a1;
    a0.x = arow0[0]; a0.y = arow0[1];
    a1.x = arow1[0]; a1.y = arow1[1];
    const float* b0 = bcol + (size_t)(k + 2 * half) * K3;
    const float* b1 = b0 + K3;
    __builtin_prefetch(arow0 + 32, 0, 0);
    __builtin_prefetch(b0 + 8 * (size_t)K3, 0, 0);
#pragma unroll
    for (int t = 0; t < 4; ++t) {
      v2f bb;
      bb.x = b0[16 * t];
      bb.y = b1[16 * t];
      acc[0][t] = __builtin_amdgcn_wmma_f32_16x16x4_f32(
          false, a0, false, bb, (short)0, acc[0][t], false, false);
      acc[1][t] = __builtin_amdgcn_wmma_f32_16x16x4_f32(
          false, a1, false, bb, (short)0, acc[1][t], false, false);
    }
    arow0 += 4;
    arow1 += 4;
  }

#pragma unroll
  for (int g = 0; g < 2; ++g) {
#pragma unroll
    for (int t = 0; t < 4; ++t) {
#pragma unroll
      for (int p = 0; p < 8; ++p) {
        int row = m0 + g * 16 + p + 8 * half;  // token index in [0, B*S)
        int col = n0 + 16 * t + lr;            // qkv column in [0, 3*NX)
        float val = acc[g][t][p] + bias[col];
        int s = row & (S_ - 1);
        int bb = row >> 11;
        if (col < NX_) {
          int h = col >> 6, d = col & 63;
          qws[(((size_t)bb * NH_ + h) * S_ + s) * HD_ + d] = val;
        } else if (col < 2 * NX_) {
          int c = col - NX_;
          int h = c >> 6, d = c & 63;
          kout[(((size_t)bb * NH_ + h) * S_ + s) * HD_ + d] = val;
        } else {
          int c = col - 2 * NX_;
          int h = c >> 6, d = c & 63;
          vout[(((size_t)bb * NH_ + h) * S_ + s) * HD_ + d] = val;
        }
      }
    }
  }
}

// ---------------------------------------------------------------------------
// Kernel C: causal flash attention, bf16 WMMA (16x16x32), fp32 softmax/accum.
// Block = 4 waves x 16 q-rows; key blocks of 32. 1/sqrt(HD) folded into Q.
// V tiles are double-buffered in LDS via async global->LDS copies
// (ASYNCcnt pipeline) when available.
// ---------------------------------------------------------------------------
__global__ __launch_bounds__(128)
void attn(const float* __restrict__ qws, const float* __restrict__ kp,
          const float* __restrict__ vp, float* __restrict__ aout) {
  __shared__ float  Vraw[2][32][HD_];  // raw fp32 V tiles: [buf][key][dim]
  __shared__ __bf16 Pl[4][16][34];     // per-wave P tile: [q row][local key]

  const int bh = blockIdx.x;           // b*NH + h
  const int qb = blockIdx.y;           // 64-row q block
  const int b  = bh >> 4;
  const int h  = bh & 15;
  const int tid  = threadIdx.x;
  const int w    = tid >> 5;
  const int lane = tid & 31;
  const int half = lane >> 4, lr = lane & 15;
  const int q0 = qb * 64 + w * 16;

  const float* Q = qws + (size_t)bh * S_ * HD_;
  const float* K = kp  + (size_t)bh * S_ * HD_;
  const float* V = vp  + (size_t)bh * S_ * HD_;

  const int nkb = 2 * qb + 2;          // cover keys [0, qb*64+64) (causal)

  // stage one 32-key V block into LDS buffer (ib & 1)
  auto stage = [&](int ib) {
    if (ib >= nkb) return;
    const int kb = ib * 32;
    const int bufi = ib & 1;
#pragma unroll
    for (int j = 0; j < 4; ++j) {
      int lin = (tid + 128 * j) * 4;   // float index within 32x64 tile
      int key = lin >> 6, dim = lin & 63;
#if defined(HAVE_ASYNC_LDS)
      __builtin_amdgcn_global_load_async_to_lds_b128(
          (gv4i_p)(V + (size_t)(kb + key) * HD_ + dim),
          (lv4i_p)&Vraw[bufi][key][dim],
          0, 0);
#else
      *(float4*)&Vraw[bufi][key][dim] =
          *(const float4*)(V + (size_t)(kb + key) * HD_ + dim);
#endif
    }
  };

  // Q A-fragments (dims 0-31 and 32-63), pre-scaled by 1/sqrt(64)
  v16bf aq0, aq1;
  const float* qrow = Q + (size_t)(q0 + lr) * HD_;
#pragma unroll
  for (int e = 0; e < 16; ++e) {
    int kk = ((e & 8) ? 16 : 0) + 8 * half + (e & 7);
    aq0[e] = (__bf16)(qrow[kk] * 0.125f);
    aq1[e] = (__bf16)(qrow[32 + kk] * 0.125f);
  }

  v8f o[4] = {};
  float mrow[8], lrow[8];
#pragma unroll
  for (int p = 0; p < 8; ++p) { mrow[p] = -1e30f; lrow[p] = 0.f; }

  stage(0);                            // prologue: block 0 in flight

  for (int ib = 0; ib < nkb; ++ib) {
    const int kb = ib * 32;
    const int bufi = ib & 1;
    __syncthreads();                   // prior readers of buf[(ib+1)&1] done
    stage(ib + 1);                     // overlap next copy with this block's math
#if defined(HAVE_ASYNC_LDS)
    if (ib + 1 < nkb) {
      WAIT_ASYNC(4);                   // block ib's copies landed; next in flight
    } else {
      WAIT_ASYNC(0);
    }
#endif
    __syncthreads();

    // K B-fragments: two 16-key score tiles, each needs dims 0-31 / 32-63
    const float* kr0 = K + (size_t)(kb + lr) * HD_;
    const float* kr1 = kr0 + (size_t)16 * HD_;
    v16bf b0a, b0b, b1a, b1b;
#pragma unroll
    for (int e = 0; e < 16; ++e) {
      int d = ((e & 8) ? 16 : 0) + 8 * half + (e & 7);
      b0a[e] = (__bf16)kr0[d];
      b0b[e] = (__bf16)kr0[32 + d];
      b1a[e] = (__bf16)kr1[d];
      b1b[e] = (__bf16)kr1[32 + d];
    }
    v8f c0 = {}, c1 = {};
    c0 = __builtin_amdgcn_wmma_f32_16x16x32_bf16(false, aq0, false, b0a, (short)0, c0, false, false);
    c0 = __builtin_amdgcn_wmma_f32_16x16x32_bf16(false, aq1, false, b0b, (short)0, c0, false, false);
    c1 = __builtin_amdgcn_wmma_f32_16x16x32_bf16(false, aq0, false, b1a, (short)0, c1, false, false);
    c1 = __builtin_amdgcn_wmma_f32_16x16x32_bf16(false, aq1, false, b1b, (short)0, c1, false, false);

    // online softmax update (row = p + 8*half within wave tile)
    float alpha[8];
#pragma unroll
    for (int p = 0; p < 8; ++p) {
      int row = q0 + p + 8 * half;
      float s0 = (kb + lr      <= row) ? c0[p] : -1e30f;
      float s1 = (kb + 16 + lr <= row) ? c1[p] : -1e30f;
      float mx = fmaxf(s0, s1);
      mx = fmaxf(mx, __shfl_xor(mx, 1, 32));
      mx = fmaxf(mx, __shfl_xor(mx, 2, 32));
      mx = fmaxf(mx, __shfl_xor(mx, 4, 32));
      mx = fmaxf(mx, __shfl_xor(mx, 8, 32));
      float mn = fmaxf(mrow[p], mx);
      float al = __expf(mrow[p] - mn);
      float p0 = __expf(s0 - mn);
      float p1 = __expf(s1 - mn);
      float rs = p0 + p1;
      rs += __shfl_xor(rs, 1, 32);
      rs += __shfl_xor(rs, 2, 32);
      rs += __shfl_xor(rs, 4, 32);
      rs += __shfl_xor(rs, 8, 32);
      lrow[p] = lrow[p] * al + rs;
      mrow[p] = mn;
      alpha[p] = al;
      Pl[w][p + 8 * half][lr]      = (__bf16)p0;
      Pl[w][p + 8 * half][16 + lr] = (__bf16)p1;
    }
#pragma unroll
    for (int t = 0; t < 4; ++t)
#pragma unroll
      for (int p = 0; p < 8; ++p) o[t][p] *= alpha[p];
    __syncthreads();

    // P A-fragment from LDS, then O += P @ V (4 dim tiles of 16)
    v16bf ap;
#pragma unroll
    for (int e = 0; e < 16; ++e) {
      int kk = ((e & 8) ? 16 : 0) + 8 * half + (e & 7);
      ap[e] = Pl[w][lr][kk];
    }
#pragma unroll
    for (int t = 0; t < 4; ++t) {
      v16bf bv;
#pragma unroll
      for (int e = 0; e < 16; ++e)
        bv[e] = (__bf16)Vraw[bufi][((e & 8) ? 16 : 0) + 8 * half + (e & 7)][t * 16 + lr];
      o[t] = __builtin_amdgcn_wmma_f32_16x16x32_bf16(false, ap, false, bv, (short)0, o[t], false, false);
    }
  }

  // normalize and write attention output in (B, S, NX) layout for the proj GEMM
#pragma unroll
  for (int p = 0; p < 8; ++p) {
    int s = q0 + p + 8 * half;
    float inv = 1.0f / lrow[p];
    size_t base = ((size_t)b * S_ + s) * NX_ + h * HD_;
#pragma unroll
    for (int t = 0; t < 4; ++t) aout[base + t * 16 + lr] = o[t][p] * inv;
  }
}

// ---------------------------------------------------------------------------
// Kernel D: output projection, fp32 WMMA (16x16x4), out = A @ W_proj + b_proj
// Wave tile = 32 rows x 64 cols.
// ---------------------------------------------------------------------------
__global__ __launch_bounds__(256)
void proj_gemm(const float* __restrict__ A, const float* __restrict__ Wp,
               const float* __restrict__ bias, float* __restrict__ out) {
  const int lane = threadIdx.x & 31;
  const int warp = threadIdx.x >> 5;
  const int half = lane >> 4, lr = lane & 15;
  const int m0 = blockIdx.y * 64 + (warp >> 2) * 32;
  const int n0 = blockIdx.x * 256 + (warp & 3) * 64;

  v8f acc[2][4] = {};
  const float* arow0 = A + (size_t)(m0 + lr) * NX_ + 2 * half;
  const float* arow1 = arow0 + (size_t)16 * NX_;
  const float* bcol  = Wp + n0 + lr;

  for (int k = 0; k < NX_; k += 4) {
    v2f a0, a1;
    a0.x = arow0[0]; a0.y = arow0[1];
    a1.x = arow1[0]; a1.y = arow1[1];
    const float* b0 = bcol + (size_t)(k + 2 * half) * NX_;
    const float* b1 = b0 + NX_;
    __builtin_prefetch(arow0 + 32, 0, 0);
    __builtin_prefetch(b0 + 8 * (size_t)NX_, 0, 0);
#pragma unroll
    for (int t = 0; t < 4; ++t) {
      v2f bb;
      bb.x = b0[16 * t];
      bb.y = b1[16 * t];
      acc[0][t] = __builtin_amdgcn_wmma_f32_16x16x4_f32(
          false, a0, false, bb, (short)0, acc[0][t], false, false);
      acc[1][t] = __builtin_amdgcn_wmma_f32_16x16x4_f32(
          false, a1, false, bb, (short)0, acc[1][t], false, false);
    }
    arow0 += 4;
    arow1 += 4;
  }

#pragma unroll
  for (int g = 0; g < 2; ++g) {
#pragma unroll
    for (int t = 0; t < 4; ++t) {
#pragma unroll
      for (int p = 0; p < 8; ++p) {
        int row = m0 + g * 16 + p + 8 * half;
        int col = n0 + 16 * t + lr;
        out[(size_t)row * NX_ + col] = acc[g][t][p] + bias[col];
      }
    }
  }
}

// ---------------------------------------------------------------------------
extern "C" void kernel_launch(void* const* d_in, const int* in_sizes, int n_in,
                              void* d_out, int out_size, void* d_ws, size_t ws_size,
                              hipStream_t stream) {
  const float* x   = (const float*)d_in[0];
  const float* Wa  = (const float*)d_in[1];
  const float* ba  = (const float*)d_in[2];
  const float* lA  = (const float*)d_in[3];
  const float* lB  = (const float*)d_in[4];
  const float* Wp  = (const float*)d_in[5];
  const float* bp  = (const float*)d_in[6];

  float* out  = (float*)d_out;               // (B, S, NX)
  float* kout = out + OUT_ELEMS;             // present[0]: (B, NH, S, HD)
  float* vout = kout + PRES;                 // present[1]

  float* ws   = (float*)d_ws;
  float* Weff = ws;                          // NX * 3NX floats
  float* qws  = Weff + (size_t)NX_ * K3;     // B*NH*S*HD floats
  float* aws  = qws + PRES;                  // B*S*NX floats

  build_weff<<<dim3((NX_ * K3 + 255) / 256), 256, 0, stream>>>(Wa, lA, lB, Weff);
  qkv_gemm<<<dim3(K3 / 256, MROWS / 64), 256, 0, stream>>>(x, Weff, ba, qws, kout, vout);
  attn<<<dim3(B_ * NH_, S_ / 64), 128, 0, stream>>>(qws, kout, vout, aws);
  proj_gemm<<<dim3(NX_ / 256, MROWS / 64), 256, 0, stream>>>(aws, Wp, bp, out);
}
